// FFSP_Decoder_75557064671559
// MI455X (gfx1250) — matrix operand order repaired and checked by reading, not compiled, not run on hardware
//
#include <hip/hip_runtime.h>

#define B_     1024
#define J_     511
#define JP_    512
#define E_     256
#define TROWS  16            // rows per subtile
#define NST    32            // 512 rows / 16
#define BPITCH 132           // dwords per bf16 LDS row (128 + 4 pad: 2-way conflicts max)

typedef __attribute__((ext_vector_type(16))) __bf16       v16bf;
typedef __attribute__((ext_vector_type(8)))  float        v8f;
typedef __attribute__((ext_vector_type(8)))  unsigned int v8u;
typedef __attribute__((ext_vector_type(4)))  unsigned int v4u;
typedef __attribute__((ext_vector_type(8)))  int          v8i;
typedef __attribute__((ext_vector_type(4)))  int          v4i;

#if defined(__AMDGCN__) && __has_builtin(__builtin_amdgcn_tensor_load_to_lds) && \
    __has_builtin(__builtin_amdgcn_s_wait_tensorcnt)
#define USE_TDM 1
#else
#define USE_TDM 0
#endif

// round-to-nearest-even fp32 -> bf16, packed pair into one dword
static __device__ __forceinline__ unsigned int pack2bf(float lo, float hi) {
  unsigned int a = __float_as_uint(lo);
  unsigned int b = __float_as_uint(hi);
  a += 0x7fffu + ((a >> 16) & 1u);
  b += 0x7fffu + ((b >> 16) & 1u);
  return (a >> 16) | (b & 0xffff0000u);
}

// ---------------------------------------------------------------------------
// Kernel 1: pack Wk/Wv into bf16 WMMA B-fragments. Frag(mat, h, kt): lane L
// holds W[h*16 + L%16][kt*32 + 16*(L/16) + 0..15] (16 contiguous bf16 = 32B).
// ---------------------------------------------------------------------------
__global__ __launch_bounds__(256) void pack_weights(
    const float* __restrict__ Wk, const float* __restrict__ Wv,
    unsigned int* __restrict__ packed) {
  int tid  = blockIdx.x * 256 + threadIdx.x;   // 8192 total
  int lane = tid & 31;
  int kt   = (tid >> 5) & 7;
  int h    = (tid >> 8) & 15;
  int mat  = (tid >> 12) & 1;
  const float* W = mat ? Wv : Wk;
  int n  = h * 16 + (lane & 15);
  int c0 = kt * 32 + ((lane >> 4) << 4);
  const float* src = W + n * E_ + c0;
  v8u o;
#pragma unroll
  for (int i = 0; i < 8; ++i) o[i] = pack2bf(src[2 * i], src[2 * i + 1]);
  ((v8u*)packed)[(((mat * 16 + h) * 8 + kt) << 5) + lane] = o;
}

// ---------------------------------------------------------------------------
// Kernel 2: q[b][e] = (machine_b . Wq[e]) * (1/sqrt(D))
// ---------------------------------------------------------------------------
__global__ __launch_bounds__(256) void q_prep(
    const float* __restrict__ machine, const float* __restrict__ Wq,
    float* __restrict__ qout) {
  __shared__ float xm[E_];
  int b = blockIdx.x, e = threadIdx.x;
  xm[e] = machine[(size_t)b * E_ + e];
  __syncthreads();
  const float* w = Wq + e * E_;
  float s = 0.f;
#pragma unroll 8
  for (int k = 0; k < E_; ++k) s += xm[k] * w[k];
  qout[(size_t)b * E_ + e] = s * 0.25f;   // 1/sqrt(16)
}

// online-softmax update straight from WMMA D-layout fragments.
// kacc[r] @ lane L = K[r + 8*(L/16)][d = L%16]; same rows for vacc.
static __device__ __forceinline__ void online_update(
    const v8f& kacc, const v8f& vacc, float q, const float* mrow, int g,
    float& m, float& l, float& o) {
  float s[8];
#pragma unroll
  for (int r = 0; r < 8; ++r) {
    float p = q * kacc[r];
    p += __shfl_xor(p, 1);
    p += __shfl_xor(p, 2);
    p += __shfl_xor(p, 4);
    p += __shfl_xor(p, 8);            // sum over the 16 lanes of this half
    s[r] = p + mrow[r + 8 * g];       // + ninf_mask
  }
  float tm = s[0];
#pragma unroll
  for (int r = 1; r < 8; ++r) tm = fmaxf(tm, s[r]);
  tm = fmaxf(tm, __shfl_xor(tm, 16));
  float mn = fmaxf(m, tm);
  float sc = __expf(m - mn);
  o *= sc; l *= sc;
#pragma unroll
  for (int r = 0; r < 8; ++r) {
    float p = __expf(s[r] - mn);
    l += p;
    o += p * vacc[r];                 // P rows line up with V-frag rows
  }
  m = mn;
}

#if USE_TDM
// Tensor DMA: 2-D tile (256 x nrows fp32) -> LDS, per CDNA5 D# layout (ISA §8)
static __device__ __forceinline__ void tdm_fetch(
    const float* gsrc, unsigned int lds_off, int nrows) {
  unsigned long long ga = (unsigned long long)(size_t)gsrc;
  v4u g0 = { 1u,                                    // count=1, user descriptor
             lds_off,                               // lds_addr (bytes)
             (unsigned int)ga,                      // global_addr[31:0]
             ((unsigned int)(ga >> 32) & 0x01ffffffu) | 0x80000000u }; // [56:32] | type=2
  v8i g1 = { (int)(2u << 16),                       // data_size = 4 bytes
             (int)(256u << 16),                     // tensor_dim0 = 256 (lo 16 in [31:16])
             (int)((unsigned int)nrows << 16),      // tensor_dim0 hi | tensor_dim1 = nrows
             (int)(256u << 16),                     // tensor_dim1 hi | tile_dim0 = 256
             nrows,                                 // tile_dim1 = nrows, tile_dim2 = 0
             256,                                   // tensor_dim0_stride = 256
             0, 0 };
  v4i z4 = { 0, 0, 0, 0 };
  v8i z8 = { 0, 0, 0, 0, 0, 0, 0, 0 };
  __builtin_amdgcn_tensor_load_to_lds(g0, g1, z4, z4, z8, 0);
}
#endif

// ---------------------------------------------------------------------------
// Kernel 3: fully fused decoder, one workgroup (8 waves) per batch element.
// ---------------------------------------------------------------------------
__global__ __launch_bounds__(256) void ffsp_decoder(
    const float* __restrict__ jobs, const float* __restrict__ nojob,
    const float* __restrict__ nmask, const float* __restrict__ qws,
    const unsigned int* __restrict__ wpack, const float* __restrict__ Wc,
    const float* __restrict__ bc, float* __restrict__ out) {
#if USE_TDM
  __shared__ __align__(16) float Xf[2][TROWS * E_];   // fp32 TDM double buffer
#endif
  __shared__ __align__(16) unsigned int XB[TROWS * BPITCH]; // bf16 tile
  __shared__ float mtile[TROWS];
  __shared__ __align__(16) float outc[E_];
  __shared__ __align__(16) float mh[E_];
  __shared__ float logits[JP_];
  __shared__ float red[16];

  const int b    = blockIdx.x;
  const int t    = threadIdx.x;
  const int w    = t >> 5;
  const int lane = t & 31;
  const int g    = lane >> 4;       // WMMA lane half
  const int lm   = lane & 15;

  const int h0 = 2 * w, h1 = 2 * w + 1;             // two heads per wave
  const float q0 = qws[(size_t)b * E_ + h0 * 16 + lm];
  const float q1 = qws[(size_t)b * E_ + h1 * 16 + lm];
  float m0 = -1e30f, m1 = -1e30f, l0 = 0.f, l1 = 0.f, o0 = 0.f, o1 = 0.f;

  const v8u* wp = (const v8u*)wpack;

#if USE_TDM
  if (w == 0)
    tdm_fetch(jobs + (size_t)b * J_ * E_, (unsigned int)(size_t)&Xf[0][0], TROWS);
#endif

  // ---- Pass 1: 32 subtiles of 16 rows: fused K/V projection + attention ----
  for (int st = 0; st < NST; ++st) {
#if USE_TDM
    if (w == 0) __builtin_amdgcn_s_wait_tensorcnt(0);
#endif
    __syncthreads();   // fp32 buffer ready (TDM) AND bf16 buffer free

    // stage + convert fp32 -> packed bf16 LDS tile (one conversion per elem)
#pragma unroll
    for (int i = 0; i < 4; ++i) {
      int q4  = t + (i << 8);          // float4 index in subtile (0..1023)
      int row = q4 >> 6;
      int c4  = q4 & 63;
      int j   = (st << 4) + row;
      float4 xv;
#if USE_TDM
      if (j < J_) xv = *(const float4*)&Xf[st & 1][row * E_ + (c4 << 2)];
      else        xv = ((const float4*)nojob)[c4];
#else
      const float4* src = (j < J_)
          ? (const float4*)(jobs + ((size_t)b * J_ + j) * E_)
          : (const float4*)nojob;
      xv = src[c4];
      if (j + TROWS < J_)
        __builtin_prefetch(
            (const void*)((const float4*)(jobs + ((size_t)b * J_ + j + TROWS) * E_) + c4),
            0, 1);
#endif
      uint2 pk;
      pk.x = pack2bf(xv.x, xv.y);
      pk.y = pack2bf(xv.z, xv.w);
      *(uint2*)&XB[row * BPITCH + (c4 << 1)] = pk;
    }
    if (t < TROWS) mtile[t] = nmask[(size_t)b * JP_ + (st << 4) + t];
#if USE_TDM
    if (w == 0 && st + 1 < NST) {      // prefetch next subtile via TDM
      int j0 = (st + 1) << 4;
      int nr = (J_ - j0 < TROWS) ? (J_ - j0) : TROWS;
      tdm_fetch(jobs + ((size_t)b * J_ + j0) * E_,
                (unsigned int)(size_t)&Xf[(st + 1) & 1][0], nr);
    }
#endif
    __syncthreads();   // bf16 tile ready

    // A fragments: 2x ds_load_b128 per kt, contiguous k = 16*g + 0..15 order, same as packed B
    v16bf A[8];
    {
      const unsigned int* xr = &XB[lm * BPITCH + (g << 3)];
#pragma unroll
      for (int kt = 0; kt < 8; ++kt) {
        v4u a0 = *(const v4u*)&xr[kt * 16];
        v4u a1 = *(const v4u*)&xr[kt * 16 + 4];
        A[kt] = __builtin_bit_cast(
            v16bf, __builtin_shufflevector(a0, a1, 0, 1, 2, 3, 4, 5, 6, 7));
      }
    }
    v8f kacc[2], vacc[2];
#pragma unroll
    for (int nt = 0; nt < 2; ++nt) {
      int h = h0 + nt;
      v8f ak = {};
      v8f av = {};
#pragma unroll
      for (int kt = 0; kt < 8; ++kt) {
        v16bf Bk = __builtin_bit_cast(v16bf, wp[(((0 * 16 + h) * 8 + kt) << 5) + lane]);
        ak = __builtin_amdgcn_wmma_f32_16x16x32_bf16(
            false, A[kt], false, Bk, (short)0, ak, false, false);
      }
#pragma unroll
      for (int kt = 0; kt < 8; ++kt) {
        v16bf Bv = __builtin_bit_cast(v16bf, wp[(((1 * 16 + h) * 8 + kt) << 5) + lane]);
        av = __builtin_amdgcn_wmma_f32_16x16x32_bf16(
            false, A[kt], false, Bv, (short)0, av, false, false);
      }
      kacc[nt] = ak; vacc[nt] = av;
    }
    online_update(kacc[0], vacc[0], q0, mtile, g, m0, l0, o0);
    online_update(kacc[1], vacc[1], q1, mtile, g, m1, l1, o1);
  }

  // ---- finalize attention ----
  l0 += __shfl_xor(l0, 16); o0 += __shfl_xor(o0, 16);
  l1 += __shfl_xor(l1, 16); o1 += __shfl_xor(o1, 16);
  if (g == 0) outc[h0 * 16 + lm] = o0 / l0;
  else        outc[h1 * 16 + lm] = o1 / l1;
  __syncthreads();

  // ---- multi_head_combine: mh = out_concat @ Wc^T + bc ----
  {
    float acc = bc[t];
    const float* wr = Wc + t * E_;
#pragma unroll 8
    for (int k = 0; k < E_; ++k) acc += outc[k] * wr[k];
    mh[t] = acc;
  }
  __syncthreads();

  // ---- Pass 2: pointer scores + tanh clip ----
  float4 mha = *(const float4*)&mh[lane * 8];
  float4 mhb = *(const float4*)&mh[lane * 8 + 4];
  for (int jj = 0; jj < 64; ++jj) {
    int j = (w << 6) + jj;
    const float4* x4 = (j < J_)
        ? (const float4*)(jobs + ((size_t)b * J_ + j) * E_)
        : (const float4*)nojob;
    float4 xa = x4[lane * 2];
    float4 xb = x4[lane * 2 + 1];
    float p = mha.x * xa.x + mha.y * xa.y + mha.z * xa.z + mha.w * xa.w +
              mhb.x * xb.x + mhb.y * xb.y + mhb.z * xb.z + mhb.w * xb.w;
    p += __shfl_xor(p, 1);  p += __shfl_xor(p, 2);  p += __shfl_xor(p, 4);
    p += __shfl_xor(p, 8);  p += __shfl_xor(p, 16);
    if (lane == 0)
      logits[j] = 10.0f * tanhf(p * 0.0625f) + nmask[(size_t)b * JP_ + j];
  }
  __syncthreads();

  // ---- final softmax over 512 ----
  float v0 = logits[t], v1 = logits[t + 256];
  float mx = fmaxf(v0, v1);
#pragma unroll
  for (int msk = 1; msk < 32; msk <<= 1) mx = fmaxf(mx, __shfl_xor(mx, msk));
  if (lane == 0) red[w] = mx;
  __syncthreads();
  float M = red[0];
#pragma unroll
  for (int i = 1; i < 8; ++i) M = fmaxf(M, red[i]);
  float e0 = __expf(v0 - M), e1 = __expf(v1 - M);
  float sm = e0 + e1;
#pragma unroll
  for (int msk = 1; msk < 32; msk <<= 1) sm += __shfl_xor(sm, msk);
  if (lane == 0) red[8 + w] = sm;
  __syncthreads();
  float S = 0.f;
#pragma unroll
  for (int i = 0; i < 8; ++i) S += red[8 + i];
  float inv = 1.0f / S;
  out[(size_t)b * JP_ + t]       = e0 * inv;
  out[(size_t)b * JP_ + 256 + t] = e1 * inv;
}

// ---------------------------------------------------------------------------
extern "C" void kernel_launch(void* const* d_in, const int* in_sizes, int n_in,
                              void* d_out, int out_size, void* d_ws, size_t ws_size,
                              hipStream_t stream) {
  const float* machine = (const float*)d_in[0];
  const float* jobs    = (const float*)d_in[1];
  const float* nmask   = (const float*)d_in[2];
  const float* Wq      = (const float*)d_in[3];
  const float* Wk      = (const float*)d_in[4];
  const float* Wv      = (const float*)d_in[5];
  const float* Wc      = (const float*)d_in[6];
  const float* bc      = (const float*)d_in[7];
  const float* nojob   = (const float*)d_in[8];

  unsigned int* wpack = (unsigned int*)d_ws;                 // 256 KB
  float*        qws   = (float*)((char*)d_ws + 256 * 1024);  // 1 MB

  pack_weights<<<32, 256, 0, stream>>>(Wk, Wv, wpack);
  q_prep<<<B_, 256, 0, stream>>>(machine, Wq, qws);
  ffsp_decoder<<<B_, 256, 0, stream>>>(jobs, nojob, nmask, qws, wpack,
                                       Wc, bc, (float*)d_out);
}
